// SpeakerEncoderLoss_7739531067889
// MI455X (gfx1250) — matrix operand order repaired and checked by reading, not compiled, not run on hardware
//
#include <hip/hip_runtime.h>
#include <hip/hip_bf16.h>

typedef __attribute__((ext_vector_type(16))) __bf16 v16bf;
typedef __attribute__((ext_vector_type(8)))  float  v8f;

#define N_SPK 2048
#define M_UTT 16
#define D_EMB 512
#define NM    (N_SPK * M_UTT)
#define EPSF  1e-8f
#define LOG2E 1.4426950408889634f

// ---- bf16 hi/lo split helpers (bit ops; round-to-nearest-even) ----
__device__ __forceinline__ unsigned short f2bf(float x) {
    unsigned int u = __float_as_uint(x);
    unsigned int r = u + 0x7FFFu + ((u >> 16) & 1u);
    return (unsigned short)(r >> 16);
}
__device__ __forceinline__ float bf2f(unsigned short h) {
    return __uint_as_float(((unsigned int)h) << 16);
}

// ============================================================
// Kernel 1: per-speaker sums + unit-clamped centroid Cn (fp32)
// grid = N_SPK, block = 256
// ============================================================
__global__ void __launch_bounds__(256)
spk_stats_kernel(const float* __restrict__ v,
                 float* __restrict__ ssum,
                 float* __restrict__ Cn) {
    const int n = blockIdx.x, tid = threadIdx.x;
    __shared__ float red[256];
    float s0 = 0.f, s1 = 0.f, p = 0.f;
    {
        const int d0 = tid, d1 = tid + 256;
        #pragma unroll
        for (int m = 0; m < M_UTT; ++m) {
            const float* row = v + ((size_t)n * M_UTT + m) * D_EMB;
            s0 += row[d0];
            s1 += row[d1];
        }
        float c0 = s0 * (1.f / M_UTT), c1 = s1 * (1.f / M_UTT);
        p = c0 * c0 + c1 * c1;
    }
    red[tid] = p; __syncthreads();
    for (int off = 128; off > 0; off >>= 1) {
        if (tid < off) red[tid] += red[tid + off];
        __syncthreads();
    }
    const float inv = 1.f / fmaxf(sqrtf(red[0]), EPSF);
    ssum[(size_t)n * D_EMB + tid]       = s0;
    ssum[(size_t)n * D_EMB + tid + 256] = s1;
    Cn[(size_t)n * D_EMB + tid]         = s0 * (1.f / M_UTT) * inv;
    Cn[(size_t)n * D_EMB + tid + 256]   = s1 * (1.f / M_UTT) * inv;
}

// ============================================================
// Kernel 2: per-utterance 1/||v||, S_diag (leave-one-out cos),
//           S_own = cos(Cn[spk], v).  grid = NM, block = 256
// ============================================================
__global__ void __launch_bounds__(256)
utt_stats_kernel(const float* __restrict__ v,
                 const float* __restrict__ ssum,
                 const float* __restrict__ Cn,
                 float* __restrict__ inv_norm,
                 float* __restrict__ S_diag,
                 float* __restrict__ S_own) {
    const int j = blockIdx.x, spk = j >> 4, tid = threadIdx.x;
    __shared__ float r0[256], r1[256], r2[256], r3[256];
    float vv = 0.f, cc = 0.f, cv = 0.f, gv = 0.f;
    #pragma unroll
    for (int e = 0; e < 2; ++e) {
        const int d = tid + e * 256;
        const float x = v[(size_t)j * D_EMB + d];
        const float s = ssum[(size_t)spk * D_EMB + d];
        const float g = Cn[(size_t)spk * D_EMB + d];
        const float cm = (s - x) * (1.f / (M_UTT - 1));
        vv += x * x; cc += cm * cm; cv += cm * x; gv += g * x;
    }
    r0[tid] = vv; r1[tid] = cc; r2[tid] = cv; r3[tid] = gv; __syncthreads();
    for (int off = 128; off > 0; off >>= 1) {
        if (tid < off) {
            r0[tid] += r0[tid + off]; r1[tid] += r1[tid + off];
            r2[tid] += r2[tid + off]; r3[tid] += r3[tid + off];
        }
        __syncthreads();
    }
    if (tid == 0) {
        const float nv = fmaxf(sqrtf(r0[0]), EPSF);
        const float nc = fmaxf(sqrtf(r1[0]), EPSF);
        inv_norm[j] = 1.f / nv;
        S_diag[j]   = r2[0] / (nc * nv);
        S_own[j]    = r3[0] / nv;   // Cn already unit-clamped
    }
}

// ============================================================
// Kernel 3: pre-swizzle Cn into WMMA A-fragment layout, bf16 hi/lo.
// A 16x32 bf16 layout (ISA 7.12.2): lanes 0-15 M=lane, K = i<8?i:i+8;
// lanes 16-31 M=lane-16, K += 8.  Per (rowtile, kchunk, lane): 16 hi + 16 lo.
// grid = 128 row tiles, block = 256
// ============================================================
__global__ void __launch_bounds__(256)
pack_A_kernel(const float* __restrict__ Cn, unsigned short* __restrict__ Apack) {
    const int tile = blockIdx.x, tid = threadIdx.x;
    for (int idx = tid; idx < 8192; idx += 256) {
        const int chunk = idx >> 9;
        const int rem   = idx & 511;
        const int lane  = rem >> 4;
        const int i     = rem & 15;
        const int Mrow  = lane & 15;
        const int K     = chunk * 32 + ((i < 8) ? i : i + 8) + ((lane & 16) ? 8 : 0);
        const float x = Cn[((size_t)tile * 16 + Mrow) * D_EMB + K];
        const unsigned short hi = f2bf(x);
        const unsigned short lo = f2bf(x - bf2f(hi));
        const size_t base = (((size_t)(tile * 16 + chunk)) * 32 + lane) * 32;
        Apack[base + i]      = hi;
        Apack[base + 16 + i] = lo;
    }
}

// ============================================================
// Kernel 4: fused GEMM + exp + column-sum.
// Each block owns 16 utterance columns; stages normalized bf16 hi/lo
// B-fragments in LDS, loops over all 2048 centroid rows with
// v_wmma_f32_16x16x32_bf16 (3-product hi/lo split), accumulates
// sum_n exp(w*S[n,j]+b) per column.  grid = NM/16, block = 256 (8 waves).
// B 32x16 bf16 layout: lanes 0-15 N=lane K = i<8?i:i+8; lanes 16-31 N=lane-16 K += 8.
// ============================================================
__global__ void __launch_bounds__(256)
gemm_lse_kernel(const float* __restrict__ v,
                const float* __restrict__ inv_norm,
                const unsigned short* __restrict__ Apack,
                const float* __restrict__ wp,
                const float* __restrict__ bp,
                float* __restrict__ sumexp) {
    __shared__ __align__(32) unsigned short Bf[16 * 32 * 32]; // 32 KB
    __shared__ float colsum[8 * 16];

    const int tid  = threadIdx.x;
    const int lane = tid & 31;
    const int wave = tid >> 5;
    const int colbase = blockIdx.x * 16;
    const float w = wp[0], b = bp[0];

    // ---- stage B tile (normalize + split + swizzle into fragment order) ----
    for (int idx = tid; idx < 8192; idx += 256) {
        const int chunk = idx >> 9;
        const int rem   = idx & 511;
        const int l     = rem >> 4;
        const int i     = rem & 15;
        const int Ncol  = l & 15;
        const int K     = chunk * 32 + ((i < 8) ? i : i + 8) + ((l & 16) ? 8 : 0);
        const int j     = colbase + Ncol;
        const float x   = v[(size_t)j * D_EMB + K] * inv_norm[j];
        const unsigned short hi = f2bf(x);
        const unsigned short lo = f2bf(x - bf2f(hi));
        const int base = (chunk * 32 + l) * 32;
        Bf[base + i]      = hi;
        Bf[base + 16 + i] = lo;
    }
    __syncthreads();

    float colpart = 0.f;
    for (int it = 0; it < 8; ++it) {
        const int t0 = it * 16 + wave * 2;      // two 16-row tiles per wave
        v8f acc0 = {}; v8f acc1 = {};
        const unsigned short* a0 = Apack + (((size_t)t0 * 16) * 32 + lane) * 32;
        const unsigned short* a1 = Apack + (((size_t)(t0 + 1) * 16) * 32 + lane) * 32;
        #pragma unroll 4
        for (int c = 0; c < 16; ++c) {
            const v16bf a0hi = *(const v16bf*)(a0 + c * 1024);
            const v16bf a0lo = *(const v16bf*)(a0 + c * 1024 + 16);
            const v16bf a1hi = *(const v16bf*)(a1 + c * 1024);
            const v16bf a1lo = *(const v16bf*)(a1 + c * 1024 + 16);
            const v16bf bhi  = *(const v16bf*)&Bf[(c * 32 + lane) * 32];
            const v16bf blo  = *(const v16bf*)&Bf[(c * 32 + lane) * 32 + 16];
            if (c < 15) {
                __builtin_prefetch(a0 + (c + 1) * 1024, 0, 1);
                __builtin_prefetch(a1 + (c + 1) * 1024, 0, 1);
            }
            acc0 = __builtin_amdgcn_wmma_f32_16x16x32_bf16(false, a0hi, false, bhi, (short)0, acc0, false, false);
            acc1 = __builtin_amdgcn_wmma_f32_16x16x32_bf16(false, a1hi, false, bhi, (short)0, acc1, false, false);
            acc0 = __builtin_amdgcn_wmma_f32_16x16x32_bf16(false, a0hi, false, blo, (short)0, acc0, false, false);
            acc1 = __builtin_amdgcn_wmma_f32_16x16x32_bf16(false, a1hi, false, blo, (short)0, acc1, false, false);
            acc0 = __builtin_amdgcn_wmma_f32_16x16x32_bf16(false, a0lo, false, bhi, (short)0, acc0, false, false);
            acc1 = __builtin_amdgcn_wmma_f32_16x16x32_bf16(false, a1lo, false, bhi, (short)0, acc1, false, false);
        }
        // exponent bounded by w+b = 5 -> no max subtraction needed
        #pragma unroll
        for (int r = 0; r < 8; ++r) {
            colpart += exp2f((acc0[r] * w + b) * LOG2E);
            colpart += exp2f((acc1[r] * w + b) * LOG2E);
        }
    }
    // C/D layout: lane l covers column N=l&15, rows split across l and l^16
    colpart += __shfl_xor(colpart, 16, 32);
    if (lane < 16) colsum[wave * 16 + lane] = colpart;
    __syncthreads();
    if (tid < 16) {
        float t = 0.f;
        #pragma unroll
        for (int wv = 0; wv < 8; ++wv) t += colsum[wv * 16 + tid];
        sumexp[colbase + tid] = t;
    }
}

// ============================================================
// Kernel 5: diagonal correction + logsumexp + total sum.
// grid = 1, block = 256
// ============================================================
__global__ void __launch_bounds__(256)
finalize_kernel(const float* __restrict__ S_diag,
                const float* __restrict__ S_own,
                const float* __restrict__ sumexp,
                const float* __restrict__ wp,
                const float* __restrict__ bp,
                float* __restrict__ out) {
    __shared__ float red[256];
    const float w = wp[0], b = bp[0];
    float acc = 0.f;
    for (int j = threadIdx.x; j < NM; j += 256) {
        const float sd = S_diag[j];
        const float so = S_own[j];
        const float se = sumexp[j] - exp2f((so * w + b) * LOG2E)
                                   + exp2f((sd * w + b) * LOG2E);
        acc += -(w * sd + b) + logf(se);
    }
    red[threadIdx.x] = acc; __syncthreads();
    for (int off = 128; off > 0; off >>= 1) {
        if (threadIdx.x < off) red[threadIdx.x] += red[threadIdx.x + off];
        __syncthreads();
    }
    if (threadIdx.x == 0) out[0] = red[0];
}

// ============================================================
extern "C" void kernel_launch(void* const* d_in, const int* in_sizes, int n_in,
                              void* d_out, int out_size, void* d_ws, size_t ws_size,
                              hipStream_t stream) {
    const float* v  = (const float*)d_in[0];   // [NM, D] fp32
    const float* wp = (const float*)d_in[1];   // scalar
    const float* bp = (const float*)d_in[2];   // scalar
    // d_in[3] = speakers (2048) — compile-time constant here

    // workspace layout (~12.6 MB)
    float* ssum          = (float*)d_ws;                           // N*D
    float* Cn            = ssum + (size_t)N_SPK * D_EMB;           // N*D
    unsigned short* Apak = (unsigned short*)(Cn + (size_t)N_SPK * D_EMB); // N*D*2 (hi+lo)
    float* invn          = (float*)(Apak + (size_t)N_SPK * D_EMB * 2);
    float* S_diag        = invn + NM;
    float* S_own         = S_diag + NM;
    float* sumexp        = S_own + NM;

    spk_stats_kernel<<<N_SPK, 256, 0, stream>>>(v, ssum, Cn);
    utt_stats_kernel<<<NM, 256, 0, stream>>>(v, ssum, Cn, invn, S_diag, S_own);
    pack_A_kernel<<<N_SPK / 16, 256, 0, stream>>>(Cn, Apak);
    gemm_lse_kernel<<<NM / 16, 256, 0, stream>>>(v, invn, Apak, wp, bp, sumexp);
    finalize_kernel<<<1, 256, 0, stream>>>(S_diag, S_own, sumexp, wp, bp, (float*)d_out);
}
